// TEMScaledDotProductAttention_21242908246651
// MI455X (gfx1250) — compile-verified, matrix operand order checked by reading, not run on hardware
//
#include <hip/hip_runtime.h>

// CDNA5 / gfx1250 fused scaled-dot-product attention with full attn output.
// B=4, H=16, S=2048, D=64.  d_out = [output B*H*S*D | attn B*H*S*S] (fp32).

typedef __attribute__((ext_vector_type(16))) _Float16 v16h;
typedef __attribute__((ext_vector_type(8)))  float    v8f;

#define SEQ     2048
#define HD      64
#define NHEAD   16
#define NBATCH  4
#define MTILE   32
#define LSTRIDE 2052   // 2048 + 4-float pad: shifts LDS banks between rows

// Load a 16x32 f16 WMMA A/B tile from a row-major f32 matrix.
// Lane mapping per ISA 7.12.2 (16-bit A 16x32): lane<16 -> K {c0..c0+7, c0+16..c0+23},
// lane>=16 -> K {c0+8..c0+15, c0+24..c0+31}; row = lane&15 (+row0).
__device__ inline v16h load_rm_tile(const float* __restrict__ base, int row0,
                                    int stride, int col0, float scale, int lane) {
  const int r  = lane & 15;
  const int kh = (lane >> 4) & 1;
  const float* p = base + (size_t)(row0 + r) * stride + col0 + kh * 8;
  v16h out;
#pragma unroll
  for (int i = 0; i < 8; ++i) out[i]     = (_Float16)(p[i]      * scale);
#pragma unroll
  for (int i = 0; i < 8; ++i) out[8 + i] = (_Float16)(p[16 + i] * scale);
  return out;
}

// Load a 32x16 f16 WMMA B tile whose columns are contiguous in memory
// (B[k][n] = V[k0+k][n0+n], V row-major with `stride` floats per row).
__device__ inline v16h load_cm_tile(const float* __restrict__ base, int k0,
                                    int stride, int n0, int lane) {
  const int n  = lane & 15;
  const int kh = (lane >> 4) & 1;
  const float* p = base + (size_t)(k0 + kh * 8) * stride + n0 + n;
  v16h out;
#pragma unroll
  for (int i = 0; i < 8; ++i) out[i]     = (_Float16)p[(size_t)i * stride];
#pragma unroll
  for (int i = 0; i < 8; ++i) out[8 + i] = (_Float16)p[(size_t)(16 + i) * stride];
  return out;
}

__global__ void
__launch_bounds__(256)
sdpa_fused_kernel(const float* __restrict__ e, const float* __restrict__ x,
                  const int* __restrict__ mask, float* __restrict__ out,
                  float* __restrict__ attn) {
  extern __shared__ float lds[];   // MTILE x LSTRIDE f32 logits / probabilities

  const int tid   = threadIdx.x;
  const int wave  = tid >> 5;
  const int lane  = tid & 31;
  const int qblk  = blockIdx.x;
  const int bh    = blockIdx.y;
  const int b     = bh >> 4;             // / NHEAD
  const int qbase = qblk * MTILE;

  const float* ehead = e + (size_t)bh * SEQ * HD;
  const float* vhead = x + (size_t)bh * SEQ * HD;
  const int*   mhead = mask + (size_t)b * SEQ * SEQ;   // mask shape [B,1,S,S]

  const int kh = (lane >> 4) & 1;   // C/D: row = r + 8*kh
  const int ln = lane & 15;         // C/D: col = ln

  // ---------------- Phase 1: masked logits -> LDS ----------------
  // Q tiles (scaled by 1/temperature = 0.125 during f16 conversion)
  v16h a0 = load_rm_tile(ehead, qbase,      HD, 0,  0.125f, lane);
  v16h a1 = load_rm_tile(ehead, qbase,      HD, 32, 0.125f, lane);
  v16h a2 = load_rm_tile(ehead, qbase + 16, HD, 0,  0.125f, lane);
  v16h a3 = load_rm_tile(ehead, qbase + 16, HD, 32, 0.125f, lane);

  for (int kt = wave; kt < SEQ / 16; kt += 8) {   // 16 key-tiles per wave
    const int kbase = kt * 16;
    // K^T tile: same lane layout as A (both index e rows over d)
    v16h b0 = load_rm_tile(ehead, kbase, HD, 0,  1.0f, lane);
    v16h b1 = load_rm_tile(ehead, kbase, HD, 32, 1.0f, lane);

    v8f c0 = {};
    c0 = __builtin_amdgcn_wmma_f32_16x16x32_f16(false, a0, false, b0, (short)0, c0, false, false);
    c0 = __builtin_amdgcn_wmma_f32_16x16x32_f16(false, a1, false, b1, (short)0, c0, false, false);
    v8f c1 = {};
    c1 = __builtin_amdgcn_wmma_f32_16x16x32_f16(false, a2, false, b0, (short)0, c1, false, false);
    c1 = __builtin_amdgcn_wmma_f32_16x16x32_f16(false, a3, false, b1, (short)0, c1, false, false);

    const int col = kbase + ln;
#pragma unroll
    for (int r = 0; r < 8; ++r) {
      const int rr = r + 8 * kh;               // row within 16-row subtile
      float v0 = c0[r];
      float v1 = c1[r];
      if (mhead[(size_t)(qbase + rr)      * SEQ + col] == 0) v0 = -60000.0f;
      if (mhead[(size_t)(qbase + 16 + rr) * SEQ + col] == 0) v1 = -60000.0f;
      lds[(rr)      * LSTRIDE + col] = v0;
      lds[(16 + rr) * LSTRIDE + col] = v1;
    }
  }
  __syncthreads();

  // ---------------- Phase 2: softmax in LDS, stream attn to HBM ----------------
  for (int sub = 0; sub < MTILE / 8; ++sub) {          // 4 rows per wave
    const int row = wave * 4 + sub;
    float* lrow = lds + row * LSTRIDE;

    float mx = -3.0e38f;
    for (int j = lane; j < SEQ; j += 32) mx = fmaxf(mx, lrow[j]);
#pragma unroll
    for (int off = 16; off > 0; off >>= 1) mx = fmaxf(mx, __shfl_xor(mx, off, 32));

    float sum = 0.0f;
    for (int j = lane; j < SEQ; j += 32) {
      const float p = __expf(lrow[j] - mx);
      lrow[j] = p;
      sum += p;
    }
#pragma unroll
    for (int off = 16; off > 0; off >>= 1) sum += __shfl_xor(sum, off, 32);
    const float inv = 1.0f / sum;

    float* arow = attn + ((size_t)bh * SEQ + qbase + row) * SEQ;
    for (int j = lane; j < SEQ; j += 32) {
      const float p = lrow[j] * inv;
      lrow[j] = p;          // keep normalized probs for phase 3
      arow[j] = p;          // coalesced HBM stream (the 1 GiB output)
    }
  }
  __syncthreads();

  // ---------------- Phase 3: O = P @ V ----------------
  const int mtile = wave >> 2;   // 0..1
  const int ntile = wave & 3;    // 0..3  (16 cols of d each)
  v8f acc = {};
#pragma unroll 4
  for (int kb = 0; kb < SEQ; kb += 32) {
    v16h ap = load_rm_tile(lds, mtile * 16, LSTRIDE, kb, 1.0f, lane);  // P from LDS
    v16h bv = load_cm_tile(vhead, kb, HD, ntile * 16, lane);           // V columns
    acc = __builtin_amdgcn_wmma_f32_16x16x32_f16(false, ap, false, bv, (short)0, acc, false, false);
  }

  float* op = out + ((size_t)bh * SEQ + qbase + mtile * 16) * HD + ntile * 16;
#pragma unroll
  for (int r = 0; r < 8; ++r)
    op[(size_t)(r + 8 * kh) * HD + ln] = acc[r];
}

extern "C" void kernel_launch(void* const* d_in, const int* in_sizes, int n_in,
                              void* d_out, int out_size, void* d_ws, size_t ws_size,
                              hipStream_t stream) {
  (void)in_sizes; (void)n_in; (void)out_size; (void)d_ws; (void)ws_size;

  const float* e    = (const float*)d_in[0];
  const float* x    = (const float*)d_in[1];
  const int*   mask = (const int*)d_in[2];

  float* out  = (float*)d_out;
  float* attn = out + (size_t)NBATCH * NHEAD * SEQ * HD;   // tuple element 1

  const size_t smem = (size_t)MTILE * LSTRIDE * sizeof(float);  // 262,656 B <= 320 KB WGP LDS
  (void)hipFuncSetAttribute((const void*)sdpa_fused_kernel,
                            hipFuncAttributeMaxDynamicSharedMemorySize, (int)smem);

  dim3 grid(SEQ / MTILE, NBATCH * NHEAD);   // 64 x 64 workgroups
  dim3 block(256);                          // 8 wave32 per workgroup
  sdpa_fused_kernel<<<grid, block, smem, stream>>>(e, x, mask, out, attn);
}
